// MultiHeadAttention_6940667150961
// MI455X (gfx1250) — matrix operand order repaired and checked by reading, not compiled, not run on hardware
//
#include <hip/hip_runtime.h>

typedef __attribute__((ext_vector_type(16))) _Float16 v16h;
typedef __attribute__((ext_vector_type(8)))  _Float16 h8;
typedef __attribute__((ext_vector_type(8)))  float    v8f;

#define B_  2
#define S_  2048
#define D_  1024
#define H_  16
#define HD_ 64

__device__ __forceinline__ v16h mk16(h8 lo, h8 hi) {
  v16h r;
#pragma unroll
  for (int i = 0; i < 8; ++i) { r[i] = lo[i]; r[i + 8] = hi[i]; }
  return r;
}

__device__ __forceinline__ v8f wmma_f16(v16h a, v16h b, v8f c) {
  // D = A(16x32) * B(32x16) + C, f32 accumulate
  return __builtin_amdgcn_wmma_f32_16x16x32_f16(false, a, false, b, (short)0, c,
                                                false, false);
}

// ---------------------------------------------------------------- conversions
__global__ void cvt_f32_f16(const float* __restrict__ in, _Float16* __restrict__ out, int n) {
  int i = blockIdx.x * 256 + threadIdx.x;
  if (i < n) out[i] = (_Float16)in[i];
}

// out[r, c] = in[c, r];  in is [cols_out, rows_out] f32, out is [rows_out, cols_out] f16
__global__ void transpose_f32_f16(const float* __restrict__ in, _Float16* __restrict__ out,
                                  int rows_out, int cols_out) {
  int i = blockIdx.x * 256 + threadIdx.x;
  if (i < rows_out * cols_out) {
    int r = i / cols_out, c = i % cols_out;
    out[i] = (_Float16)in[(long)c * rows_out + r];
  }
}

// ---------------------------------------------------------------- QKV GEMM
// A: xh [4096,1024] f16.  Bt: WqkvT [3072,1024] f16 (row n holds W[:,n]).
// One wave computes a 64x64 output tile (16 WMMA accumulators); scatter into
// Q[B,H,S,HD], K[B,H,S,HD], Vt[B,H,HD,S].
__global__ void __launch_bounds__(32, 1)
qkv_gemm(const _Float16* __restrict__ A,
         const _Float16* __restrict__ Bt,
         const float* __restrict__ bias,
         _Float16* __restrict__ Qd,
         _Float16* __restrict__ Kd,
         _Float16* __restrict__ Vt) {
  const int lane = threadIdx.x;
  const int m = lane & 15, half = lane >> 4;
  const int ab = half * 8, kb16 = half * 16;
  const int row0 = blockIdx.x * 64;
  const int col0 = blockIdx.y * 64;

  // one-time prefetch of the first chunks (no loop-carried pressure)
  __builtin_prefetch((const void*)(A + (long)(row0 + m) * D_), 0, 1);
  __builtin_prefetch((const void*)(Bt + (long)(col0 + m) * D_), 0, 1);

  v8f acc[4][4] = {};
  for (int kc = 0; kc < D_; kc += 32) {
    v16h a[4];
#pragma unroll
    for (int mt = 0; mt < 4; ++mt) {
      const _Float16* pa = A + (long)(row0 + mt * 16 + m) * D_ + kc;
      a[mt] = mk16(*(const h8*)(pa + kb16), *(const h8*)(pa + kb16 + 8));
    }
#pragma unroll
    for (int nt = 0; nt < 4; ++nt) {
      const _Float16* pb = Bt + (long)(col0 + nt * 16 + m) * D_ + kc + kb16;
      v16h b = *(const v16h*)pb;
      acc[0][nt] = wmma_f16(a[0], b, acc[0][nt]);
      acc[1][nt] = wmma_f16(a[1], b, acc[1][nt]);
      acc[2][nt] = wmma_f16(a[2], b, acc[2][nt]);
      acc[3][nt] = wmma_f16(a[3], b, acc[3][nt]);
    }
  }
#pragma unroll
  for (int mt = 0; mt < 4; ++mt) {
#pragma unroll
    for (int nt = 0; nt < 4; ++nt) {
      const int j = col0 + nt * 16 + m;      // column in [0, 3072)
      const float bj = bias[j];
      const int which = j >> 10;             // 0=q, 1=k, 2=v
      const int h  = (j >> 6) & 15;
      const int hd = j & 63;
#pragma unroll
      for (int r = 0; r < 8; ++r) {
        const int grow = row0 + mt * 16 + r + half * 8;  // token row in [0,4096)
        const int bbi = grow >> 11;
        const int s   = grow & (S_ - 1);
        const float v = acc[mt][nt][r] + bj;
        const long bh = (long)bbi * H_ + h;
        if (which == 0)      Qd[(bh * S_ + s) * HD_ + hd] = (_Float16)v;
        else if (which == 1) Kd[(bh * S_ + s) * HD_ + hd] = (_Float16)v;
        else                 Vt[(bh * HD_ + hd) * S_ + s] = (_Float16)v;  // transposed V
      }
    }
  }
}

// ---------------------------------------------------------------- flash attention
// One wave handles one 16-query tile for one (b, h). Streams keys 32 at a time.
__global__ void flash_attn(const _Float16* __restrict__ Qd,
                           const _Float16* __restrict__ Kd,
                           const _Float16* __restrict__ Vt,
                           const float* __restrict__ mask,
                           _Float16* __restrict__ Z) {
  __shared__ _Float16 pbuf[16 * 32];  // P tile staging (C-layout -> A-layout)
  const int lane = threadIdx.x;
  const int qt = blockIdx.x;   // 0..127 query tile
  const int h  = blockIdx.y;   // 0..15
  const int bb = blockIdx.z;   // 0..1
  const int m = lane & 15, half = lane >> 4;
  const int ab = half * 8, kb16 = half * 16;

  const long bh = (long)bb * H_ + h;
  const _Float16* Qbase = Qd + (bh * S_ + qt * 16) * HD_;
  const _Float16* Kb    = Kd + bh * S_ * HD_;
  const _Float16* Vb    = Vt + bh * HD_ * S_;
  const float*    mrow  = mask + ((long)bb * S_ + qt * 16) * S_;

  // Q A-fragments for K-dim (HD) chunks [0,32) and [32,64)
  const _Float16* pq = Qbase + m * HD_;
  v16h aq0 = mk16(*(const h8*)(pq + kb16),      *(const h8*)(pq + kb16 + 8));
  v16h aq1 = mk16(*(const h8*)(pq + 32 + kb16), *(const h8*)(pq + 40 + kb16));

  v8f o0 = {}, o1 = {}, o2 = {}, o3 = {};
  float rmax[8], rsum[8];
#pragma unroll
  for (int r = 0; r < 8; ++r) { rmax[r] = -1e30f; rsum[r] = 0.0f; }

  for (int kb = 0; kb < S_; kb += 32) {
    // ---- scores: two 16x16 tiles (keys kb..kb+15, kb+16..kb+31)
    v8f z = {};
    const _Float16* pk0 = Kb + (long)(kb + m) * HD_;
    v8f s0 = wmma_f16(aq0, *(const v16h*)(pk0 + kb16), z);
    s0     = wmma_f16(aq1, *(const v16h*)(pk0 + 32 + kb16), s0);
    const _Float16* pk1 = Kb + (long)(kb + 16 + m) * HD_;
    v8f s1 = wmma_f16(aq0, *(const v16h*)(pk1 + kb16), z);
    s1     = wmma_f16(aq1, *(const v16h*)(pk1 + 32 + kb16), s1);

    // ---- scale + mask
#pragma unroll
    for (int r = 0; r < 8; ++r) {
      const float* mp = mrow + (long)(r + half * 8) * S_ + kb + m;
      s0[r] = s0[r] * 0.125f - 1e9f * mp[0];
      s1[r] = s1[r] * 0.125f - 1e9f * mp[16];
    }

    // ---- online softmax (row = VGPR index, 16 lanes per row half)
#pragma unroll
    for (int r = 0; r < 8; ++r) {
      float mx = fmaxf(s0[r], s1[r]);
      mx = fmaxf(mx, __shfl_xor(mx, 1, 32));
      mx = fmaxf(mx, __shfl_xor(mx, 2, 32));
      mx = fmaxf(mx, __shfl_xor(mx, 4, 32));
      mx = fmaxf(mx, __shfl_xor(mx, 8, 32));
      float nm = fmaxf(rmax[r], mx);
      float alpha = __expf(rmax[r] - nm);
      rmax[r] = nm;
      float p0 = __expf(s0[r] - nm);
      float p1 = __expf(s1[r] - nm);
      s0[r] = p0; s1[r] = p1;
      float ps = p0 + p1;
      ps += __shfl_xor(ps, 1, 32);
      ps += __shfl_xor(ps, 2, 32);
      ps += __shfl_xor(ps, 4, 32);
      ps += __shfl_xor(ps, 8, 32);
      rsum[r] = rsum[r] * alpha + ps;
      o0[r] *= alpha; o1[r] *= alpha; o2[r] *= alpha; o3[r] *= alpha;
    }

    // ---- P: C-layout -> LDS -> A-layout f16 fragment
#pragma unroll
    for (int r = 0; r < 8; ++r) {
      pbuf[(r + half * 8) * 32 + m]      = (_Float16)s0[r];
      pbuf[(r + half * 8) * 32 + 16 + m] = (_Float16)s1[r];
    }
    v16h pf = mk16(*(const h8*)(&pbuf[m * 32 + kb16]),
                   *(const h8*)(&pbuf[m * 32 + kb16 + 8]));

    // ---- O += P(16x32) * V(32x64), V read from transposed Vt (contig along keys)
    const _Float16* pv = Vb + (long)m * S_ + kb + kb16;
    o0 = wmma_f16(pf, *(const v16h*)(pv),           o0);
    o1 = wmma_f16(pf, *(const v16h*)(pv + 16 * S_), o1);
    o2 = wmma_f16(pf, *(const v16h*)(pv + 32 * S_), o2);
    o3 = wmma_f16(pf, *(const v16h*)(pv + 48 * S_), o3);
  }

  // ---- normalize + store into Z with the reference's faithful [H,B,S,HD] flat order
  _Float16* zrow = Z + (((long)h * B_ + bb) * S_ + qt * 16) * HD_;
#pragma unroll
  for (int r = 0; r < 8; ++r) {
    float inv = 1.0f / rsum[r];
    _Float16* zp = zrow + (r + half * 8) * HD_ + m;
    zp[0]  = (_Float16)(o0[r] * inv);
    zp[16] = (_Float16)(o1[r] * inv);
    zp[32] = (_Float16)(o2[r] * inv);
    zp[48] = (_Float16)(o3[r] * inv);
  }
}

// ---------------------------------------------------------------- output GEMM
// out[4096,1024] f32 = Z[4096,1024]f16 @ Wo + bo;  Bt = WoT [1024,1024] f16
// One wave computes a 64x64 output tile.
__global__ void __launch_bounds__(32, 1)
out_gemm(const _Float16* __restrict__ A,
         const _Float16* __restrict__ Bt,
         const float* __restrict__ bias,
         float* __restrict__ out) {
  const int lane = threadIdx.x;
  const int m = lane & 15, half = lane >> 4;
  const int ab = half * 8, kb16 = half * 16;
  const int row0 = blockIdx.x * 64;
  const int col0 = blockIdx.y * 64;

  __builtin_prefetch((const void*)(A + (long)(row0 + m) * D_), 0, 1);
  __builtin_prefetch((const void*)(Bt + (long)(col0 + m) * D_), 0, 1);

  v8f acc[4][4] = {};
  for (int kc = 0; kc < D_; kc += 32) {
    v16h a[4];
#pragma unroll
    for (int mt = 0; mt < 4; ++mt) {
      const _Float16* pa = A + (long)(row0 + mt * 16 + m) * D_ + kc;
      a[mt] = mk16(*(const h8*)(pa + kb16), *(const h8*)(pa + kb16 + 8));
    }
#pragma unroll
    for (int nt = 0; nt < 4; ++nt) {
      const _Float16* pb = Bt + (long)(col0 + nt * 16 + m) * D_ + kc + kb16;
      v16h b = *(const v16h*)pb;
      acc[0][nt] = wmma_f16(a[0], b, acc[0][nt]);
      acc[1][nt] = wmma_f16(a[1], b, acc[1][nt]);
      acc[2][nt] = wmma_f16(a[2], b, acc[2][nt]);
      acc[3][nt] = wmma_f16(a[3], b, acc[3][nt]);
    }
  }
#pragma unroll
  for (int mt = 0; mt < 4; ++mt) {
#pragma unroll
    for (int nt = 0; nt < 4; ++nt) {
      const int j = col0 + nt * 16 + m;
      const float bj = bias[j];
#pragma unroll
      for (int r = 0; r < 8; ++r) {
        const int grow = row0 + mt * 16 + r + half * 8;
        out[(long)grow * D_ + j] = acc[mt][nt][r] + bj;
      }
    }
  }
}

// ---------------------------------------------------------------- launch
extern "C" void kernel_launch(void* const* d_in, const int* in_sizes, int n_in,
                              void* d_out, int out_size, void* d_ws, size_t ws_size,
                              hipStream_t stream) {
  const float* x    = (const float*)d_in[0];   // [2,2048,1024]
  const float* mask = (const float*)d_in[1];   // [2,2048,2048]
  const float* Wqkv = (const float*)d_in[2];   // [1024,3072]
  const float* bqkv = (const float*)d_in[3];   // [3072]
  const float* Wo   = (const float*)d_in[4];   // [1024,1024]
  const float* bo   = (const float*)d_in[5];   // [1024]
  float* out = (float*)d_out;                  // [2,2048,1024] f32

  _Float16* wsh   = (_Float16*)d_ws;
  _Float16* xh    = wsh;               // 4096*1024
  _Float16* WqkvT = xh    + 4194304;   // 3072*1024
  _Float16* WoT   = WqkvT + 3145728;   // 1024*1024
  _Float16* Qd    = WoT   + 1048576;   // [B,H,S,HD]
  _Float16* Kd    = Qd    + 4194304;   // [B,H,S,HD]
  _Float16* Vt    = Kd    + 4194304;   // [B,H,HD,S]
  _Float16* Z     = Vt    + 4194304;   // [H,B,S,HD] flat (faithful reshape)

  cvt_f32_f16<<<(4194304 + 255) / 256, 256, 0, stream>>>(x, xh, 4194304);
  transpose_f32_f16<<<(3145728 + 255) / 256, 256, 0, stream>>>(Wqkv, WqkvT, 3072, 1024);
  transpose_f32_f16<<<(1048576 + 255) / 256, 256, 0, stream>>>(Wo, WoT, 1024, 1024);

  qkv_gemm<<<dim3(64, 48), 32, 0, stream>>>(xh, WqkvT, bqkv, Qd, Kd, Vt);
  flash_attn<<<dim3(128, 16, 2), 32, 0, stream>>>(Qd, Kd, Vt, mask, Z);
  out_gemm<<<dim3(64, 16), 32, 0, stream>>>(Z, WoT, bo, out);
}
